// Model_16071767621701
// MI455X (gfx1250) — 8-bit matrix operand order repaired and checked by reading, compiled, not run on hardware
//
#include <hip/hip_runtime.h>

typedef __attribute__((ext_vector_type(8))) int v8i;

#define BATCH   64
#define PIXELS  50176
#define DIMV    40
#define NPAD    48
#define KCH     784
#define MT      4
#define NT      3
#define KSPLIT  14
#define CPS     (KCH / KSPLIT)

#define WMMA_IU8(A, B, C) \
    __builtin_amdgcn_wmma_i32_16x16x64_iu8(true, (A), true, (B), (C), false, false)

__global__ void prep_a(const float* __restrict__ x, int* __restrict__ apk) {
    int t = blockIdx.x * blockDim.x + threadIdx.x;
    const int NDW = MT * KCH * 32 * 8;
    if (t >= NDW) return;
    int v    = t & 7;
    int lane = (t >> 3) & 31;
    int rest = t >> 8;
    int kc   = rest % KCH;
    int mt   = rest / KCH;

    int half = lane >> 4;
    int row  = lane & 15;
    int v2   = v & 3;
    int kst  = ((v & 4) ? 32 : 0) + ((v2 >> 1) * 16) + ((v2 & 1) * 4) + half * 8;
    int b    = mt * 16 + row;
    int p0   = kc * 64 + kst;

    const float* xr = x + (size_t)b * PIXELS + p0;
    unsigned u = 0;
#pragma unroll
    for (int j = 0; j < 4; ++j) {
        signed char sv = (xr[j] >= 0.5f) ? (signed char)1 : (signed char)-1;
        u |= ((unsigned)(unsigned char)sv) << (8 * j);
    }
    apk[t] = (int)u;
}

__global__ void prep_b(const float* __restrict__ pos, int* __restrict__ bpk) {
    int t = blockIdx.x * blockDim.x + threadIdx.x;
    const int NDW = NT * KCH * 32 * 8;
    if (t >= NDW) return;
    int v    = t & 7;
    int lane = (t >> 3) & 31;
    int rest = t >> 8;
    int kc   = rest % KCH;
    int nt   = rest / KCH;

    int half = lane >> 4;
    int n    = nt * 16 + (lane & 15);
    int kst  = ((v & 4) ? 32 : 0) + (((v & 3) >> 1) * 16) + ((v & 1) * 4) + half * 8;
    int p0   = kc * 64 + kst;

    unsigned u = 0;
    if (n < DIMV) {
#pragma unroll
        for (int j = 0; j < 4; ++j) {
            float pv = pos[(size_t)(p0 + j) * DIMV + n];
            signed char sv = (pv >= 0.0f) ? (signed char)1 : (signed char)-1;
            u |= ((unsigned)(unsigned char)sv) << (8 * j);
        }
    }
    bpk[t] = (int)u;
}

__global__ void colsum(const float* __restrict__ pos, float* __restrict__ s0) {
    __shared__ float sm[256];
    int d = blockIdx.x;
    int t = threadIdx.x;
    float acc = 0.0f;
    for (int p = t; p < PIXELS; p += 256)
        acc += pos[(size_t)p * DIMV + d];
    sm[t] = acc;
    __syncthreads();
#pragma unroll
    for (int s = 128; s > 0; s >>= 1) {
        if (t < s) sm[t] += sm[t + s];
        __syncthreads();
    }
    if (t == 0) s0[d] = sm[0];
}

__global__ void zero_mdot(int* __restrict__ mdot) {
    int t = blockIdx.x * blockDim.x + threadIdx.x;
    if (t < BATCH * NPAD) mdot[t] = 0;
}

__global__ void __launch_bounds__(32)
gemm_wmma(const int* __restrict__ apk, const int* __restrict__ bpk,
          int* __restrict__ mdot) {
    int nt   = blockIdx.x;
    int mt   = blockIdx.y;
    int ks   = blockIdx.z;
    int lane = threadIdx.x;

    const v8i* __restrict__ ap =
        (const v8i*)apk + ((size_t)mt * KCH + (size_t)ks * CPS) * 32 + lane;
    const v8i* __restrict__ bp =
        (const v8i*)bpk + ((size_t)nt * KCH + (size_t)ks * CPS) * 32 + lane;

    v8i acc0 = {}, acc1 = {}, acc2 = {}, acc3 = {};

    v8i a0 = ap[0 * 32], b0 = bp[0 * 32];
    v8i a1 = ap[1 * 32], b1 = bp[1 * 32];
    v8i a2 = ap[2 * 32], b2 = bp[2 * 32];
    v8i a3 = ap[3 * 32], b3 = bp[3 * 32];
    ap += 32 * 4;
    bp += 32 * 4;

    for (int g = 1; g < CPS / 4; ++g) {
        __builtin_prefetch((const void*)(ap + 32 * 8), 0, 1);
        __builtin_prefetch((const void*)(bp + 32 * 8), 0, 1);

        v8i na0 = ap[0 * 32], nb0 = bp[0 * 32];
        v8i na1 = ap[1 * 32], nb1 = bp[1 * 32];
        v8i na2 = ap[2 * 32], nb2 = bp[2 * 32];
        v8i na3 = ap[3 * 32], nb3 = bp[3 * 32];

        acc0 = WMMA_IU8(a0, b0, acc0);
        acc1 = WMMA_IU8(a1, b1, acc1);
        acc2 = WMMA_IU8(a2, b2, acc2);
        acc3 = WMMA_IU8(a3, b3, acc3);

        a0 = na0; b0 = nb0;
        a1 = na1; b1 = nb1;
        a2 = na2; b2 = nb2;
        a3 = na3; b3 = nb3;
        ap += 32 * 4;
        bp += 32 * 4;
    }

    acc0 = WMMA_IU8(a0, b0, acc0);
    acc1 = WMMA_IU8(a1, b1, acc1);
    acc2 = WMMA_IU8(a2, b2, acc2);
    acc3 = WMMA_IU8(a3, b3, acc3);

    acc0 += acc1;
    acc2 += acc3;
    acc0 += acc2;

    int half = lane >> 4;
    int d    = nt * 16 + (lane & 15);
#pragma unroll
    for (int r = 0; r < 8; ++r) {
        int b = mt * 16 + r + half * 8;
        atomicAdd(&mdot[b * NPAD + d], acc0[r]);
    }
}

__global__ void finalize(const int* __restrict__ mdot, const float* __restrict__ s0,
                         const float* __restrict__ lev, const float* __restrict__ cw,
                         float* __restrict__ out, int n_out) {
    int t = blockIdx.x * blockDim.x + threadIdx.x;
    if (t >= n_out) return;
    int b = t / 10;
    int c = t % 10;
    float dot = 0.0f;
#pragma unroll 8
    for (int d = 0; d < DIMV; ++d) {
        float l0 = lev[d];
        float l1 = lev[DIMV + d];
        float c0 = 0.5f * (l0 + l1);
        float c1 = 0.5f * (l1 - l0);
        float summed = c0 * s0[d] + c1 * (float)mdot[b * NPAD + d];
        float enc = (summed > 0.0f) ? 1.0f : -1.0f;
        dot += enc * cw[c * DIMV + d];
    }
    out[t] = dot;
}

extern "C" void kernel_launch(void* const* d_in, const int* in_sizes, int n_in,
                              void* d_out, int out_size, void* d_ws, size_t ws_size,
                              hipStream_t stream) {
    const float* x   = (const float*)d_in[0];
    const float* pos = (const float*)d_in[1];
    const float* lev = (const float*)d_in[2];
    const float* cw  = (const float*)d_in[3];
    float* out = (float*)d_out;

    const size_t A_BYTES = (size_t)MT * KCH * 32 * 8 * 4;
    const size_t B_BYTES = (size_t)NT * KCH * 32 * 8 * 4;
    char* ws    = (char*)d_ws;
    int*   apk  = (int*)ws;
    int*   bpk  = (int*)(ws + A_BYTES);
    float* s0   = (float*)(ws + A_BYTES + B_BYTES);
    int*   mdot = (int*)(ws + A_BYTES + B_BYTES + 256);

    const int NA = MT * KCH * 32 * 8;
    const int NB = NT * KCH * 32 * 8;

    prep_a<<<(NA + 255) / 256, 256, 0, stream>>>(x, apk);
    prep_b<<<(NB + 255) / 256, 256, 0, stream>>>(pos, bpk);
    colsum<<<DIMV, 256, 0, stream>>>(pos, s0);
    zero_mdot<<<(BATCH * NPAD + 255) / 256, 256, 0, stream>>>(mdot);

    dim3 gg(NT, MT, KSPLIT);
    gemm_wmma<<<gg, 32, 0, stream>>>(apk, bpk, mdot);

    finalize<<<(out_size + 255) / 256, 256, 0, stream>>>(mdot, s0, lev, cw, out, out_size);
}
